// AngleRegressorSharedFaces_83502754169059
// MI455X (gfx1250) — matrix operand order repaired and checked by reading, not compiled, not run on hardware
//
#include <hip/hip_runtime.h>
#include <hip/hip_bf16.h>
#include <stdint.h>

// ---------------------------------------------------------------------------
// MI455X (gfx1250, wave32) implementation.
// WMMA f32_16x16x32_f16 used for: conv2 (batch-as-M per-pixel GEMM, K=taps*ch),
// hex layer-2 GEMMs (B*73 x 64 @ 64x64), and head fc1 (1024x3072 @ 3072x256).
// Wave-uniform control flow is forced into SGPRs via readfirstlane; conv2
// weights are staged in LDS to avoid VGPR spills.
// ---------------------------------------------------------------------------

typedef __attribute__((ext_vector_type(16))) _Float16 v16h;
typedef __attribute__((ext_vector_type(8)))  _Float16 v8h;
typedef __attribute__((ext_vector_type(8)))  float    v8f;

#define BTOT 1024
#define BC   256          // batch chunk for conv path (keeps x1 in L2)
#define NPHO_W 4760
#define PIX_TOTAL 7620    // 4092 inner + 144 us + 144 ds + 3240 outer
#define NHEX 73
#define EPSBN 1e-5f

__constant__ int FACE_H[4]     = {93, 24, 24, 45};
__constant__ int FACE_W[4]     = {44,  6,  6, 72};
__constant__ int FACE_PXOFF[4] = {0, 4092, 4236, 4380};
// chunk-of-8-pixels bases per face: counts 512,18,18,405 (total 953)
#define CHUNKS_TOTAL 953

// OUTER_CENTER transposed to (5,6), hardcoded from the reference
__constant__ int OCT[5][6] = {
  {4185,4744,4194,4203,4753,4212},
  {4742,4745,4749,4751,4754,4758},
  {4186,4746,4195,4204,4755,4213},
  {4743,4747,4750,4752,4756,4759},
  {4187,4748,4196,4205,4757,4214}};

__device__ __forceinline__ float leakyf(float x){ return x > 0.f ? x : 0.1f * x; }

__device__ __forceinline__ v16h make_v16h(v8h lo, v8h hi){
  v16h r;
#pragma unroll
  for (int i = 0; i < 8; ++i){ r[i] = lo[i]; r[i+8] = hi[i]; }
  return r;
}

__device__ __forceinline__ v8f wmma16(v16h a, v16h b, v8f c){
  // D = A(16x32 f16) x B(32x16 f16) + C(16x16 f32)
  return __builtin_amdgcn_wmma_f32_16x16x32_f16(false, a, false, b, (short)0, c, false, false);
}

// ---------------------------------------------------------------------------
// prep kernels: fold BN, build per-lane WMMA B-fragments, build Anorm
// ---------------------------------------------------------------------------
__global__ void zero_f32_kernel(float* p, int n){
  int i = blockIdx.x * 256 + threadIdx.x;
  if (i < n) p[i] = 0.f;
}

__global__ void prep_small_kernel(const float* c1w, const float* c1b,
                                  const float* bn1g, const float* bn1b, const float* bn1m, const float* bn1v,
                                  const float* c2b, const float* bn2g, const float* bn2b, const float* bn2m, const float* bn2v,
                                  const float* deg,
                                  float* w1f, float* b1f, float* bias2f, float* fvec){
  int t = threadIdx.x;
  if (t < 144){ int ch = t / 9; float s = bn1g[ch] * rsqrtf(bn1v[ch] + EPSBN); w1f[t] = c1w[t] * s; }
  if (t < 16){ float s = bn1g[t] * rsqrtf(bn1v[t] + EPSBN); b1f[t] = c1b[t] * s + bn1b[t] - bn1m[t] * s; }
  if (t < 32){ float s = bn2g[t] * rsqrtf(bn2v[t] + EPSBN); bias2f[t] = c2b[t] * s + bn2b[t] - bn2m[t] * s; }
  if (t < NHEX){ fvec[t] = deg[t] > 0.f ? 1.f : 0.f; }
}

// conv2 weight frags: layout [nt(2)][ks(5)][lane(32)][e(16)] halves.
// B(32x16) 16-bit layout: lane half h holds K=8h..8h+7 (e<8) and K=16+8h..16+8h+7 (e>=8).
__global__ void prep_wfrag2_kernel(const float* c2w, const float* bn2g, const float* bn2v, _Float16* wfrag2){
  int i = blockIdx.x * 256 + threadIdx.x;
  if (i >= 2 * 5 * 32 * 16) return;
  int e = i & 15, lane = (i >> 4) & 31, rest = i >> 9;
  int ks = rest % 5, nt = rest / 5;
  int k = ks * 32 + (e >> 3) * 16 + (lane >> 4) * 8 + (e & 7);   // global K in [0,160)
  int n = nt * 16 + (lane & 15);            // out channel
  int tap = k >> 4, cin = k & 15;
  float v = 0.f;
  if (tap < 9){
    float s = bn2g[n] * rsqrtf(bn2v[n] + EPSBN);
    v = c2w[(n * 16 + cin) * 9 + tap] * s;  // (32,16,3,3) row-major, tap=dy*3+dx
  }
  wfrag2[i] = (_Float16)v;
}

// head fc1 frags: layout [ks(96)][nt(16)][lane(32)][e(16)]
__global__ void prep_whead_kernel(const float* hd1w, _Float16* whead){
  int i = blockIdx.x * 256 + threadIdx.x;
  if (i >= 96 * 16 * 32 * 16) return;
  int e = i & 15, lane = (i >> 4) & 31, nt = (i >> 9) & 15, ks = i >> 13;
  int k = ks * 32 + (lane >> 4) * 16 + e;
  int col = nt * 16 + (lane & 15);
  whead[i] = (_Float16)hd1w[k * 256 + col];  // (3072,256)
}

// hex 64x64 frags: layout [ks(2)][nt(4)][lane(32)][e(16)]
__global__ void prep_whex_kernel(const float* W, _Float16* frag){
  int i = blockIdx.x * 256 + threadIdx.x;
  if (i >= 2 * 4 * 32 * 16) return;
  int e = i & 15, lane = (i >> 4) & 31, nt = (i >> 9) & 3, ks = i >> 11;
  int k = ks * 32 + (lane >> 4) * 16 + e;
  int col = nt * 16 + (lane & 15);
  frag[i] = (_Float16)W[k * 64 + col];       // (64,64)
}

__global__ void anorm_build_kernel(const int* ei, int E, float* Anorm){
  int e = blockIdx.x * 256 + threadIdx.x;
  if (e >= E) return;
  int s = ei[e], d = ei[E + e];
  atomicAdd(&Anorm[d * NHEX + s], 1.0f);
}

__global__ void anorm_scale_kernel(const float* deg, float* Anorm){
  int i = blockIdx.x * 256 + threadIdx.x;
  if (i >= NHEX * NHEX) return;
  int n = i / NHEX;
  Anorm[i] *= 1.0f / fmaxf(deg[n], 1.0f);
}

// ---------------------------------------------------------------------------
// conv1 (1->16, BN+leaky folded) with fused face gather; writes f16
// x1 layout: [pixel_global][b_local(BC)][ch(16)]
// ---------------------------------------------------------------------------
__global__ void conv1_kernel(const float* __restrict__ npho, const float* __restrict__ w1f,
                             const float* __restrict__ b1f, _Float16* __restrict__ x1, int b0){
  int p  = (blockIdx.x % 30) * 256 + threadIdx.x;
  int bl = blockIdx.x / 30;
  if (p >= PIX_TOTAL) return;
  int face, pf;
  if (p < 4092){ face = 0; pf = p; }
  else if (p < 4236){ face = 1; pf = p - 4092; }
  else if (p < 4380){ face = 2; pf = p - 4236; }
  else { face = 3; pf = p - 4380; }
  int H = FACE_H[face], W = FACE_W[face];
  int r = pf / W, c = pf % W;
  const float* row = npho + (size_t)(b0 + bl) * NPHO_W;

  float acc[16];
#pragma unroll
  for (int j = 0; j < 16; ++j) acc[j] = b1f[j];
#pragma unroll
  for (int t = 0; t < 9; ++t){
    int rr = r + t / 3 - 1, cc = c + t % 3 - 1;
    float v = 0.f;
    if (rr >= 0 && rr < H && cc >= 0 && cc < W){
      if (face == 0)      v = row[rr * 44 + cc];
      else if (face == 1) v = row[4308 + rr * 6 + cc];
      else if (face == 2) v = row[4452 + rr * 6 + cc];
      else {
        if (rr >= 15 && rr < 30 && cc >= 30 && cc < 42)
          v = row[OCT[(rr - 15) / 3][(cc - 30) / 2]] * (1.f / 6.f);
        else
          v = row[4092 + (rr / 5) * 24 + (cc / 3)] * (1.f / 15.f);
      }
    }
#pragma unroll
    for (int j = 0; j < 16; ++j) acc[j] = fmaf(w1f[j * 9 + t], v, acc[j]);
  }
  _Float16* dst = x1 + ((size_t)p * BC + bl) * 16;
  v8h o0, o1;
#pragma unroll
  for (int j = 0; j < 8; ++j){
    o0[j] = (_Float16)leakyf(acc[j]);
    o1[j] = (_Float16)leakyf(acc[8 + j]);
  }
  *(v8h*)dst = o0;
  *(v8h*)(dst + 8) = o1;
}

// ---------------------------------------------------------------------------
// conv2 via WMMA: per wave = (face, 8-pixel chunk, 16-batch tile).
// M=batch(16), N=32 (two 16-wide C frags), K=160 (9 taps*16ch + zero pad).
// Weights staged in LDS; wave-uniform decode forced into SGPRs.
// Fused BN+bias+leaky + adaptive-4x4-pool (overlapping-bin aware) atomics.
// ---------------------------------------------------------------------------
__global__ __launch_bounds__(256) void conv2_kernel(
    const _Float16* __restrict__ x1, const _Float16* __restrict__ wfrag2,
    const float* __restrict__ bias2f, float* __restrict__ zf32, int b0){
  __shared__ _Float16 wlds[2 * 5 * 32 * 16];   // 10 KB
  {
    const uint32_t* src = (const uint32_t*)wfrag2;
    uint32_t* dst = (uint32_t*)wlds;
    for (int i = threadIdx.x; i < (2 * 5 * 32 * 16) / 2; i += 256) dst[i] = src[i];
  }
  __syncthreads();

  int lane = threadIdx.x & 31;
  // force wave-uniform decode into SGPRs -> scalar branches everywhere below
  int gw = __builtin_amdgcn_readfirstlane(blockIdx.x * 8 + (threadIdx.x >> 5));
  const int NW = CHUNKS_TOTAL * (BC / 16);
  if (gw >= NW) return;
  int btile = gw & 15;
  int chunk = gw >> 4;
  int face, cl;
  if (chunk < 512){ face = 0; cl = chunk; }
  else if (chunk < 530){ face = 1; cl = chunk - 512; }
  else if (chunk < 548){ face = 2; cl = chunk - 530; }
  else { face = 3; cl = chunk - 548; }
  int H = FACE_H[face], W = FACE_W[face];
  int pxoff = FACE_PXOFF[face];
  int g = lane >> 4;
  int laneoff = (btile * 16 + (lane & 15)) * 16 + g * 8;  // halves into pixel slab
  int nn = lane & 15;
  float bz0 = bias2f[nn];
  float bz1 = bias2f[16 + nn];

  for (int i = 0; i < 8; ++i){
    int p = cl * 8 + i;
    if (p >= H * W) break;                  // uniform
    int r = p / W, c = p % W;
    v8f acc0 = {}, acc1 = {};
#pragma unroll
    for (int ks = 0; ks < 5; ++ks){
      int t0 = ks * 2, t1 = ks * 2 + 1;
      v8h lo; { v8h z = {}; lo = z; }
      v8h hi; { v8h z = {}; hi = z; }
      {
        int rr = r + t0 / 3 - 1, cc = c + t0 % 3 - 1;
        if (rr >= 0 && rr < H && cc >= 0 && cc < W)  // uniform predicate
          lo = *(const v8h*)(x1 + (size_t)(pxoff + rr * W + cc) * (BC * 16) + laneoff);
      }
      if (t1 < 9){
        int rr = r + t1 / 3 - 1, cc = c + t1 % 3 - 1;
        if (rr >= 0 && rr < H && cc >= 0 && cc < W)  // uniform predicate
          hi = *(const v8h*)(x1 + (size_t)(pxoff + rr * W + cc) * (BC * 16) + laneoff);
      }
      v16h a = make_v16h(lo, hi);
      const _Float16* w0 = wlds + ((0 * 5 + ks) * 32 + lane) * 16;
      const _Float16* w1 = wlds + ((1 * 5 + ks) * 32 + lane) * 16;
      v16h b0v = make_v16h(*(const v8h*)w0, *(const v8h*)(w0 + 8));
      v16h b1v = make_v16h(*(const v8h*)w1, *(const v8h*)(w1 + 8));
      acc0 = wmma16(a, b0v, acc0);
      acc1 = wmma16(a, b1v, acc1);
    }

    // adaptive pool bins containing (r,c); torch bins can overlap (all uniform)
    int rb[2], rcn[2], nrh = 0;
    int cbn[2], ccn[2], nch = 0;
#pragma unroll
    for (int bi = 0; bi < 4; ++bi){
      int r0 = (bi * H) >> 2, r1 = ((bi + 1) * H + 3) >> 2;
      if (r >= r0 && r < r1 && nrh < 2){ rb[nrh] = bi; rcn[nrh] = r1 - r0; ++nrh; }
      int c0 = (bi * W) >> 2, c1 = ((bi + 1) * W + 3) >> 2;
      if (c >= c0 && c < c1 && nch < 2){ cbn[nch] = bi; ccn[nch] = c1 - c0; ++nch; }
    }

#pragma unroll
    for (int rr8 = 0; rr8 < 8; ++rr8){
      int babs = b0 + btile * 16 + rr8 + 8 * g;      // C-frag row: m = rr8 + 8*g
      float v0 = leakyf(acc0[rr8] + bz0);
      float v1 = leakyf(acc1[rr8] + bz1);
      float* zrow = zf32 + (size_t)babs * 3072 + face * 512;
      for (int ri = 0; ri < nrh; ++ri)
        for (int ci = 0; ci < nch; ++ci){
          float wgt = 1.0f / (float)(rcn[ri] * ccn[ci]);
          int slot = rb[ri] * 4 + cbn[ci];
          atomicAdd(zrow + nn * 16 + slot,        v0 * wgt);
          atomicAdd(zrow + (16 + nn) * 16 + slot, v1 * wgt);
        }
    }
  }
}

// ---------------------------------------------------------------------------
// hex encoder kernels
// ---------------------------------------------------------------------------
__global__ void hex1_kernel(const float* __restrict__ npho, const float* __restrict__ Anorm,
                            const float* __restrict__ fvec,
                            const float* h1sw, const float* h1sb, const float* h1nw, const float* h1nb,
                            _Float16* __restrict__ h1out, int idx_base){
  int b = blockIdx.x;
  int n = threadIdx.x;
  __shared__ float xs[NHEX];
  if (n < NHEX) xs[n] = npho[(size_t)b * NPHO_W + idx_base + n];
  __syncthreads();
  if (n >= NHEX) return;
  float y = 0.f;
  const float* An = Anorm + n * NHEX;
  for (int m = 0; m < NHEX; ++m) y = fmaf(An[m], xs[m], y);
  float x = xs[n], fn = fvec[n];
  _Float16* o = h1out + (size_t)(b * NHEX + n) * 64;
  for (int c = 0; c < 64; ++c){
    float h = leakyf(x * h1sw[c] + h1sb[c] + y * h1nw[c] + fn * h1nb[c]);
    o[c] = (_Float16)h;
  }
}

// (B*73 x 64) @ (64 x 64) -> f16 out, via WMMA. MT = 74752/16 = 4672.
__global__ __launch_bounds__(256) void gemm64_kernel(
    const _Float16* __restrict__ A, const _Float16* __restrict__ Wfrag, _Float16* __restrict__ Out){
  int lane = threadIdx.x & 31;
  int gw = __builtin_amdgcn_readfirstlane(blockIdx.x * 8 + (threadIdx.x >> 5));
  const int MT = (BTOT * NHEX) / 16;
  if (gw >= MT * 4) return;
  int nt = gw & 3;
  int mt = gw >> 2;
  int g = lane >> 4;
  int mload = mt * 16 + (lane & 15);
  v8f acc = {};
#pragma unroll
  for (int ks = 0; ks < 2; ++ks){
    const _Float16* ap = A + (size_t)mload * 64 + ks * 32 + g * 16;
    v16h a = make_v16h(*(const v8h*)ap, *(const v8h*)(ap + 8));
    const _Float16* wp = Wfrag + ((ks * 4 + nt) * 32 + lane) * 16;
    v16h b = make_v16h(*(const v8h*)wp, *(const v8h*)(wp + 8));
    acc = wmma16(a, b, acc);
  }
  int ncol = nt * 16 + (lane & 15);
#pragma unroll
  for (int rr = 0; rr < 8; ++rr){
    int row = mt * 16 + rr + 8 * g;
    Out[(size_t)row * 64 + ncol] = (_Float16)acc[rr];
  }
}

// agg = Anorm @ t ; h2 = leaky(u + sb2 + agg + f*nb2)
__global__ void hex2_kernel(const float* __restrict__ Anorm, const float* __restrict__ fvec,
                            const _Float16* __restrict__ tbuf, const _Float16* __restrict__ ubuf,
                            const float* h2sb, const float* h2nb, _Float16* __restrict__ h2out){
  int i = blockIdx.x * 256 + threadIdx.x;
  if (i >= BTOT * NHEX * 64) return;
  int ch = i & 63;
  int bn = i >> 6;
  int b = bn / NHEX, n = bn % NHEX;
  float agg = 0.f;
  const _Float16* tb = tbuf + (size_t)b * NHEX * 64 + ch;
  const float* An = Anorm + n * NHEX;
  for (int m = 0; m < NHEX; ++m) agg = fmaf(An[m], (float)tb[(size_t)m * 64], agg);
  float u = (float)ubuf[(size_t)bn * 64 + ch];
  float h = leakyf(u + h2sb[ch] + agg + fvec[n] * h2nb[ch]);
  h2out[(size_t)bn * 64 + ch] = (_Float16)h;
}

__global__ void hex_mean_kernel(const _Float16* __restrict__ h2, float* __restrict__ hm){
  int i = blockIdx.x * 256 + threadIdx.x;
  if (i >= BTOT * 64) return;
  int ch = i & 63, b = i >> 6;
  float s = 0.f;
  const _Float16* p = h2 + (size_t)b * NHEX * 64 + ch;
  for (int n = 0; n < NHEX; ++n) s += (float)p[(size_t)n * 64];
  hm[i] = s * (1.0f / 73.0f);
}

__global__ void hex_q_kernel(const float* __restrict__ hm, const float* __restrict__ p1w,
                             const float* __restrict__ p1b, float* __restrict__ q){
  int i = blockIdx.x * 256 + threadIdx.x;
  if (i >= BTOT * 64) return;
  int c = i & 63, b = i >> 6;
  float acc = p1b[c];
  const float* h = hm + b * 64;
  for (int k = 0; k < 64; ++k) acc = fmaf(h[k], p1w[k * 64 + c], acc);
  q[i] = leakyf(acc);
}

__global__ void hex_emb_kernel(const float* __restrict__ q, const float* __restrict__ p2w,
                               const float* __restrict__ p2b, float* __restrict__ zf32, int eoff){
  int i = blockIdx.x * 256 + threadIdx.x;
  if (i >= BTOT * 512) return;
  int j = i & 511, b = i >> 9;
  float acc = p2b[j];
  const float* qb = q + b * 64;
  for (int k = 0; k < 64; ++k) acc = fmaf(qb[k], p2w[k * 512 + j], acc);
  zf32[(size_t)b * 3072 + eoff + j] = acc;
}

// ---------------------------------------------------------------------------
// head
// ---------------------------------------------------------------------------
__global__ void convert_zh_kernel(const float* __restrict__ zf32, _Float16* __restrict__ zh){
  int i = blockIdx.x * 256 + threadIdx.x;
  if (i < BTOT * 3072) zh[i] = (_Float16)zf32[i];
}

// (1024 x 3072) @ (3072 x 256), K-split 4 across waves, atomic reduce into f32
__global__ __launch_bounds__(256) void head1_kernel(
    const _Float16* __restrict__ zh, const _Float16* __restrict__ whead, float* __restrict__ hidden){
  int lane = threadIdx.x & 31;
  int gw = __builtin_amdgcn_readfirstlane(blockIdx.x * 8 + (threadIdx.x >> 5));
  if (gw >= 64 * 16 * 4) return;
  int split = gw & 3;
  int nt = (gw >> 2) & 15;
  int mt = gw >> 6;
  int g = lane >> 4;
  int mload = mt * 16 + (lane & 15);
  v8f acc = {};
  for (int ks = split * 24; ks < split * 24 + 24; ++ks){
    const _Float16* ap = zh + (size_t)mload * 3072 + ks * 32 + g * 16;
    v16h a = make_v16h(*(const v8h*)ap, *(const v8h*)(ap + 8));
    const _Float16* wp = whead + (((size_t)ks * 16 + nt) * 32 + lane) * 16;
    v16h b = make_v16h(*(const v8h*)wp, *(const v8h*)(wp + 8));
    acc = wmma16(a, b, acc);
  }
  int ncol = nt * 16 + (lane & 15);
#pragma unroll
  for (int rr = 0; rr < 8; ++rr){
    int row = mt * 16 + rr + 8 * g;
    atomicAdd(&hidden[(size_t)row * 256 + ncol], acc[rr]);
  }
}

__global__ void head2_kernel(const float* __restrict__ hidden, const float* __restrict__ hd1b,
                             const float* __restrict__ hd2w, const float* __restrict__ hd2b,
                             float* __restrict__ out){
  int b = blockIdx.x * 256 + threadIdx.x;
  if (b >= BTOT) return;
  float a0 = hd2b[0], a1 = hd2b[1];
  const float* h = hidden + (size_t)b * 256;
  for (int j = 0; j < 256; ++j){
    float hv = leakyf(h[j] + hd1b[j]);
    a0 = fmaf(hv, hd2w[j * 2 + 0], a0);
    a1 = fmaf(hv, hd2w[j * 2 + 1], a1);
  }
  out[b * 2 + 0] = a0;
  out[b * 2 + 1] = a1;
}

// ---------------------------------------------------------------------------
extern "C" void kernel_launch(void* const* d_in, const int* in_sizes, int n_in,
                              void* d_out, int out_size, void* d_ws, size_t ws_size,
                              hipStream_t stream){
  const float* npho = (const float*)d_in[0];
  const float* c1w  = (const float*)d_in[1];
  const float* c1b  = (const float*)d_in[2];
  const float* bn1g = (const float*)d_in[3];
  const float* bn1b = (const float*)d_in[4];
  const float* bn1m = (const float*)d_in[5];
  const float* bn1v = (const float*)d_in[6];
  const float* c2w  = (const float*)d_in[7];
  const float* c2b  = (const float*)d_in[8];
  const float* bn2g = (const float*)d_in[9];
  const float* bn2b = (const float*)d_in[10];
  const float* bn2m = (const float*)d_in[11];
  const float* bn2v = (const float*)d_in[12];
  const float* h1sw = (const float*)d_in[13];
  const float* h1sb = (const float*)d_in[14];
  const float* h1nw = (const float*)d_in[15];
  const float* h1nb = (const float*)d_in[16];
  const float* h2sw = (const float*)d_in[17];
  const float* h2sb = (const float*)d_in[18];
  const float* h2nw = (const float*)d_in[19];
  const float* h2nb = (const float*)d_in[20];
  const float* p1w  = (const float*)d_in[21];
  const float* p1b  = (const float*)d_in[22];
  const float* p2w  = (const float*)d_in[23];
  const float* p2b  = (const float*)d_in[24];
  const float* hd1w = (const float*)d_in[25];
  const float* hd1b = (const float*)d_in[26];
  const float* hd2w = (const float*)d_in[27];
  const float* hd2b = (const float*)d_in[28];
  const float* deg  = (const float*)d_in[29];
  const int*   ei   = (const int*)d_in[30];
  int E = in_sizes[30] / 2;
  float* out = (float*)d_out;

  // workspace carve-out (256B aligned)
  uint8_t* wsp = (uint8_t*)d_ws;
  auto carve = [&](size_t bytes) -> void* {
    void* p = (void*)wsp;
    wsp += (bytes + 255) & ~(size_t)255;
    return p;
  };
  _Float16* x1     = (_Float16*)carve((size_t)PIX_TOTAL * BC * 16 * 2);   // ~62 MB
  float*    zf32   = (float*)   carve((size_t)BTOT * 3072 * 4);
  _Float16* zh     = (_Float16*)carve((size_t)BTOT * 3072 * 2);
  _Float16* bufH   = (_Float16*)carve((size_t)BTOT * NHEX * 64 * 2);
  _Float16* tbuf   = (_Float16*)carve((size_t)BTOT * NHEX * 64 * 2);
  _Float16* ubuf   = (_Float16*)carve((size_t)BTOT * NHEX * 64 * 2);
  float*    hm     = (float*)   carve((size_t)BTOT * 64 * 4);
  float*    qb     = (float*)   carve((size_t)BTOT * 64 * 4);
  float*    hidden = (float*)   carve((size_t)BTOT * 256 * 4);
  float*    Anorm  = (float*)   carve((size_t)NHEX * NHEX * 4);
  float*    fvec   = (float*)   carve((size_t)NHEX * 4);
  float*    w1f    = (float*)   carve(144 * 4);
  float*    b1f    = (float*)   carve(16 * 4);
  float*    bias2f = (float*)   carve(32 * 4);
  _Float16* wfrag2 = (_Float16*)carve((size_t)2 * 5 * 32 * 16 * 2);
  _Float16* whead  = (_Float16*)carve((size_t)96 * 16 * 32 * 16 * 2);
  _Float16* wn2    = (_Float16*)carve((size_t)2 * 4 * 32 * 16 * 2);
  _Float16* ws2    = (_Float16*)carve((size_t)2 * 4 * 32 * 16 * 2);
  (void)ws_size; (void)n_in; (void)out_size;

  // 1) zero accumulators (every call: deterministic)
  zero_f32_kernel<<<(BTOT * 3072 + 255) / 256, 256, 0, stream>>>(zf32, BTOT * 3072);
  zero_f32_kernel<<<(BTOT * 256 + 255) / 256, 256, 0, stream>>>(hidden, BTOT * 256);
  zero_f32_kernel<<<(NHEX * NHEX + 255) / 256, 256, 0, stream>>>(Anorm, NHEX * NHEX);

  // 2) prep
  prep_small_kernel<<<1, 256, 0, stream>>>(c1w, c1b, bn1g, bn1b, bn1m, bn1v,
                                           c2b, bn2g, bn2b, bn2m, bn2v, deg,
                                           w1f, b1f, bias2f, fvec);
  prep_wfrag2_kernel<<<(2 * 5 * 32 * 16 + 255) / 256, 256, 0, stream>>>(c2w, bn2g, bn2v, wfrag2);
  prep_whead_kernel<<<(96 * 16 * 32 * 16 + 255) / 256, 256, 0, stream>>>(hd1w, whead);
  prep_whex_kernel<<<(2 * 4 * 32 * 16 + 255) / 256, 256, 0, stream>>>(h2nw, wn2);
  prep_whex_kernel<<<(2 * 4 * 32 * 16 + 255) / 256, 256, 0, stream>>>(h2sw, ws2);
  anorm_build_kernel<<<(E + 255) / 256, 256, 0, stream>>>(ei, E, Anorm);
  anorm_scale_kernel<<<(NHEX * NHEX + 255) / 256, 256, 0, stream>>>(deg, Anorm);

  // 3) conv backbones: batch chunks of 256 (x1 chunk fits in L2)
  for (int ch = 0; ch < BTOT / BC; ++ch){
    int b0 = ch * BC;
    conv1_kernel<<<30 * BC, 256, 0, stream>>>(npho, w1f, b1f, x1, b0);
    int conv2_waves = CHUNKS_TOTAL * (BC / 16);
    conv2_kernel<<<(conv2_waves * 32 + 255) / 256, 256, 0, stream>>>(x1, wfrag2, bias2f, zf32, b0);
  }

  // 4) hex encoders (top, bottom)
  for (int e = 0; e < 2; ++e){
    int idx_base = 4596 + e * NHEX;
    hex1_kernel<<<BTOT, 128, 0, stream>>>(npho, Anorm, fvec, h1sw, h1sb, h1nw, h1nb, bufH, idx_base);
    int g64_waves = ((BTOT * NHEX) / 16) * 4;
    gemm64_kernel<<<(g64_waves * 32 + 255) / 256, 256, 0, stream>>>(bufH, wn2, tbuf);
    gemm64_kernel<<<(g64_waves * 32 + 255) / 256, 256, 0, stream>>>(bufH, ws2, ubuf);
    hex2_kernel<<<(BTOT * NHEX * 64 + 255) / 256, 256, 0, stream>>>(Anorm, fvec, tbuf, ubuf, h2sb, h2nb, bufH);
    hex_mean_kernel<<<(BTOT * 64 + 255) / 256, 256, 0, stream>>>(bufH, hm);
    hex_q_kernel<<<(BTOT * 64 + 255) / 256, 256, 0, stream>>>(hm, p1w, p1b, qb);
    hex_emb_kernel<<<(BTOT * 512 + 255) / 256, 256, 0, stream>>>(qb, p2w, p2b, zf32, 2048 + e * 512);
  }

  // 5) head
  convert_zh_kernel<<<(BTOT * 3072 + 255) / 256, 256, 0, stream>>>(zf32, zh);
  head1_kernel<<<(64 * 16 * 4 * 32 + 255) / 256, 256, 0, stream>>>(zh, whead, hidden);
  head2_kernel<<<(BTOT + 255) / 256, 256, 0, stream>>>(hidden, hd1b, hd2w, hd2b, out);
}
